// CrossAttention_32341103739428
// MI455X (gfx1250) — matrix operand order repaired and checked by reading, not compiled, not run on hardware
//
#include <hip/hip_runtime.h>
#include <hip/hip_bf16.h>

// ---------------------------------------------------------------------------
// CDNA5 (gfx1250) cross-attention:  bf16 WMMA everywhere, f32 accumulate.
// B=4, Nq=Nk=2048, C=768, H=12, Dh=64.
// f32->bf16 conversion hoisted to a one-shot pre-pass; GEMM/attention inner
// loops are pure b128 loads + v_wmma. 32-row wave tiles double B-fragment
// reuse (loads per WMMA: proj 1.5, attention 1.0).
// ---------------------------------------------------------------------------

typedef __attribute__((ext_vector_type(16))) __bf16          v16bf;
typedef __attribute__((ext_vector_type(8)))  float           v8f;
typedef __attribute__((ext_vector_type(8)))  unsigned short  v8us;

union Frag16 {                 // 16 bf16 values = 8 VGPRs (A or B operand)
    v16bf          v;
    v8us           h[2];       // two 16-byte chunks
    unsigned short s[16];
};

__device__ __forceinline__ unsigned short f2bf(float f) {
    union { float f; unsigned int u; } c; c.f = f;
    unsigned int u = c.u;
    u += 0x7FFFu + ((u >> 16) & 1u);          // round-to-nearest-even
    return (unsigned short)(u >> 16);
}

__device__ __forceinline__ v8f wmma_bf16(const Frag16& a, const Frag16& b, v8f c) {
    return __builtin_amdgcn_wmma_f32_16x16x32_bf16(
        /*neg_a=*/false, a.v, /*neg_b=*/false, b.v,
        /*c_mod=*/(short)0, c, /*reuse_a=*/false, /*reuse_b=*/false);
}

#define NHEAD 12
#define SEQ   2048
#define CDIM  768
#define DH    64

// ---------------------------------------------------------------------------
// One-shot f32 -> bf16 conversion (4 elements / thread, b128 in, b64 out).
// ---------------------------------------------------------------------------
__global__ __launch_bounds__(256)
void ca_cvt_bf16(const float* __restrict__ src, unsigned short* __restrict__ dst, int n4)
{
    const int i = blockIdx.x * blockDim.x + threadIdx.x;
    if (i < n4) {
        const float4 v = ((const float4*)src)[i];
        union { unsigned short s[4]; unsigned long long u; } p;
        p.s[0] = f2bf(v.x); p.s[1] = f2bf(v.y); p.s[2] = f2bf(v.z); p.s[3] = f2bf(v.w);
        ((unsigned long long*)dst)[i] = p.u;
    }
}

// ---------------------------------------------------------------------------
// WMMA projection GEMM:  out = A(bf16)[rows x 768] * W(bf16)[768 x 768]^T
// mode 0: bf16 out, head-split layout [b][h][n][d]   (Q with scale, K)
// mode 1: bf16 out, transposed head layout [b][h][d][n]  (V)
// mode 2: f32 out + bias, row-major [rows][768]          (final projection)
// Block = 128 threads (4 waves). Block tile 128(M) x 64(N); wave = 32(M) x 64(N).
// Inner loop: 12 global_load_b128 + 8 v_wmma (every B fragment feeds 2 WMMA).
// ---------------------------------------------------------------------------
__global__ __launch_bounds__(128)
void ca_proj_wmma(const unsigned short* __restrict__ A, const unsigned short* __restrict__ W,
                  void* __restrict__ out, const float* __restrict__ bias,
                  int mode, float scale)
{
    const int lane = threadIdx.x & 31;
    const int wave = threadIdx.x >> 5;
    const int half = lane >> 4;          // 0: lanes 0-15, 1: lanes 16-31
    const int lid  = lane & 15;
    const int mbase = blockIdx.x * 128 + wave * 32;
    const int nbase = blockIdx.y * 64;

    const v8us* arow0 = (const v8us*)(A + (size_t)(mbase +  0 + lid) * CDIM);
    const v8us* arow1 = (const v8us*)(A + (size_t)(mbase + 16 + lid) * CDIM);
    const v8us* wrow0 = (const v8us*)(W + (size_t)(nbase +  0 + lid) * CDIM);
    const v8us* wrow1 = (const v8us*)(W + (size_t)(nbase + 16 + lid) * CDIM);
    const v8us* wrow2 = (const v8us*)(W + (size_t)(nbase + 32 + lid) * CDIM);
    const v8us* wrow3 = (const v8us*)(W + (size_t)(nbase + 48 + lid) * CDIM);

    v8f acc[2][4];
#pragma unroll
    for (int g = 0; g < 2; ++g)
#pragma unroll
        for (int s = 0; s < 4; ++s) acc[g][s] = (v8f){0.f,0.f,0.f,0.f,0.f,0.f,0.f,0.f};

    for (int k = 0; k < CDIM; k += 32) {
        // A fragments (16x32 each): rows mbase(+16)+lid, K = k + 16h + 0..15 contiguous
        const int i0 = (k >> 3) + half * 2;
        Frag16 a0; a0.h[0] = arow0[i0]; a0.h[1] = arow0[i0 + 1];
        Frag16 a1; a1.h[0] = arow1[i0]; a1.h[1] = arow1[i0 + 1];
        // B fragments (32x16): element (K,N) = W[N][K]; K = k + 16h + 0..15 contiguous
        const int j0 = (k >> 3) + half * 2;
        Frag16 b0; b0.h[0] = wrow0[j0]; b0.h[1] = wrow0[j0 + 1];
        acc[0][0] = wmma_bf16(a0, b0, acc[0][0]);
        acc[1][0] = wmma_bf16(a1, b0, acc[1][0]);
        Frag16 b1; b1.h[0] = wrow1[j0]; b1.h[1] = wrow1[j0 + 1];
        acc[0][1] = wmma_bf16(a0, b1, acc[0][1]);
        acc[1][1] = wmma_bf16(a1, b1, acc[1][1]);
        Frag16 b2; b2.h[0] = wrow2[j0]; b2.h[1] = wrow2[j0 + 1];
        acc[0][2] = wmma_bf16(a0, b2, acc[0][2]);
        acc[1][2] = wmma_bf16(a1, b2, acc[1][2]);
        Frag16 b3; b3.h[0] = wrow3[j0]; b3.h[1] = wrow3[j0 + 1];
        acc[0][3] = wmma_bf16(a0, b3, acc[0][3]);
        acc[1][3] = wmma_bf16(a1, b3, acc[1][3]);
    }

    // epilogue: D layout — element r: row = mbase + g*16 + r + 8*half, col = nbase+s*16+lid
#pragma unroll
    for (int g = 0; g < 2; ++g) {
#pragma unroll
        for (int s = 0; s < 4; ++s) {
            const int c = nbase + s * 16 + lid;
#pragma unroll
            for (int r = 0; r < 8; ++r) {
                const int   R   = mbase + g * 16 + r + 8 * half;   // flat row = b*SEQ + n
                const float val = acc[g][s][r] * scale;
                if (mode == 0) {
                    const int b = R >> 11, q = R & (SEQ - 1), h = c >> 6, d = c & (DH - 1);
                    ((unsigned short*)out)[((((size_t)b * NHEAD + h) * SEQ + q) << 6) + d] = f2bf(val);
                } else if (mode == 1) {
                    const int b = R >> 11, q = R & (SEQ - 1), h = c >> 6, d = c & (DH - 1);
                    ((unsigned short*)out)[((((size_t)b * NHEAD + h) << 6) + d) * SEQ + q] = f2bf(val);
                } else {
                    ((float*)out)[(size_t)R * CDIM + c] = val + bias[c];
                }
            }
        }
    }
}

// ---------------------------------------------------------------------------
// Flash attention: one block = one (b,h) x 128 Q rows; 4 waves x 32-row strips
// (two 16-row groups per wave, sharing every K/V fragment).
// Qb, Kb: bf16 [b][h][n][64]   Vt: bf16 [b][h][64][n]   Xo: bf16 [b][n][768]
// Scale (Dh^-0.5) is pre-folded into Qb by the projection kernel.
// ---------------------------------------------------------------------------
#define PSTR 40   // LDS row stride in halves (80 B: 16B-aligned, bank-skewed)

__global__ __launch_bounds__(128)
void ca_flash_attn_wmma(const unsigned short* __restrict__ Qb,
                        const unsigned short* __restrict__ Kb,
                        const unsigned short* __restrict__ Vt,
                        unsigned short* __restrict__ Xo)
{
    __shared__ unsigned short Plds[4 * 32 * PSTR];   // per-wave P relayout scratch

    const int lane = threadIdx.x & 31;
    const int wave = threadIdx.x >> 5;
    const int half = lane >> 4;
    const int lid  = lane & 15;
    const int bh   = blockIdx.y;                 // 0..47
    const int b    = bh / NHEAD, h = bh % NHEAD;
    const int qrow = blockIdx.x * 128 + wave * 32;

    // Q A-fragments (kept in registers): 2 row groups x 2 k-steps over Dh=64
    Frag16 qa[2][2];
#pragma unroll
    for (int g = 0; g < 2; ++g) {
        const v8us* qr = (const v8us*)(Qb + (((size_t)bh * SEQ) + qrow + g * 16 + lid) * DH);
        qa[g][0].h[0] = qr[2 * half];      qa[g][0].h[1] = qr[2 * half + 1];   // d = 16h .. 16h+15
        qa[g][1].h[0] = qr[4 + 2 * half];  qa[g][1].h[1] = qr[5 + 2 * half];   // d = 32+16h .. 32+16h+15
    }

    v8f o[2][4];
#pragma unroll
    for (int g = 0; g < 2; ++g)
#pragma unroll
        for (int s = 0; s < 4; ++s) o[g][s] = (v8f){0.f,0.f,0.f,0.f,0.f,0.f,0.f,0.f};
    float mst[2][8], lst[2][8];
#pragma unroll
    for (int g = 0; g < 2; ++g)
#pragma unroll
        for (int r = 0; r < 8; ++r) { mst[g][r] = -1e30f; lst[g][r] = 0.f; }

    unsigned short* pl = Plds + wave * (32 * PSTR);

    for (int kt = 0; kt < SEQ; kt += 32) {
        // ------------- S = Q K^T  (32q x 32k), 8 WMMA, 8 loads -------------
        v8f sg[2][2];
#pragma unroll
        for (int g = 0; g < 2; ++g) {
            sg[g][0] = (v8f){0.f,0.f,0.f,0.f,0.f,0.f,0.f,0.f};
            sg[g][1] = (v8f){0.f,0.f,0.f,0.f,0.f,0.f,0.f,0.f};
        }
        {
            const v8us* kr0 = (const v8us*)(Kb + ((size_t)bh * SEQ + kt + lid) * DH);
            const v8us* kr1 = (const v8us*)(Kb + ((size_t)bh * SEQ + kt + 16 + lid) * DH);
#pragma unroll
            for (int dd = 0; dd < 2; ++dd) {                  // d = dd*32 + 16h .. +15
                Frag16 kbf;
                kbf.h[0] = kr0[dd * 4 + half * 2]; kbf.h[1] = kr0[dd * 4 + half * 2 + 1];
                sg[0][0] = wmma_bf16(qa[0][dd], kbf, sg[0][0]);
                sg[1][0] = wmma_bf16(qa[1][dd], kbf, sg[1][0]);
                kbf.h[0] = kr1[dd * 4 + half * 2]; kbf.h[1] = kr1[dd * 4 + half * 2 + 1];
                sg[0][1] = wmma_bf16(qa[0][dd], kbf, sg[0][1]);
                sg[1][1] = wmma_bf16(qa[1][dd], kbf, sg[1][1]);
            }
        }
        // ------------- online softmax (rows live in 16-lane halves) -------------
#pragma unroll
        for (int g = 0; g < 2; ++g) {
#pragma unroll
            for (int r = 0; r < 8; ++r) {
                float v = fmaxf(sg[g][0][r], sg[g][1][r]);
#pragma unroll
                for (int msk = 1; msk < 16; msk <<= 1) v = fmaxf(v, __shfl_xor(v, msk, 16));
                const float mnew  = fmaxf(mst[g][r], v);
                const float alpha = __expf(mst[g][r] - mnew);
                mst[g][r] = mnew;
                const float p0 = __expf(sg[g][0][r] - mnew);
                const float p1 = __expf(sg[g][1][r] - mnew);
                sg[g][0][r] = p0; sg[g][1][r] = p1;
                float ps = p0 + p1;
#pragma unroll
                for (int msk = 1; msk < 16; msk <<= 1) ps += __shfl_xor(ps, msk, 16);
                lst[g][r] = lst[g][r] * alpha + ps;
                o[g][0][r] *= alpha; o[g][1][r] *= alpha;
                o[g][2][r] *= alpha; o[g][3][r] *= alpha;
            }
        }
        // ------------- P: D-frag -> LDS -> A-frag relayout -------------
#pragma unroll
        for (int g = 0; g < 2; ++g) {
#pragma unroll
            for (int r = 0; r < 8; ++r) {
                const int row = g * 16 + r + 8 * half;          // D layout: M = r + 8*half
                pl[row * PSTR + lid]      = f2bf(sg[g][0][r]);
                pl[row * PSTR + 16 + lid] = f2bf(sg[g][1][r]);
            }
        }
        __syncthreads();
        Frag16 pa[2];
#pragma unroll
        for (int g = 0; g < 2; ++g) {
            const v8us* pr = (const v8us*)(pl + (g * 16 + lid) * PSTR); // A layout: row = lid
            pa[g].h[0] = pr[2 * half];                          // kcol = 16*half ..
            pa[g].h[1] = pr[2 * half + 1];                      // kcol = 16*half+8 ..
        }
        // ------------- O += P V  (32q x 64d), 8 WMMA, 8 loads -------------
#pragma unroll
        for (int s = 0; s < 4; ++s) {
            Frag16 vb;
            const v8us* vr = (const v8us*)(Vt + (((size_t)bh * DH) + s * 16 + lid) * SEQ);
            const int j = (kt >> 3) + half * 2;                 // kcol = kt + 16*half ..
            vb.h[0] = vr[j]; vb.h[1] = vr[j + 1];
            o[0][s] = wmma_bf16(pa[0], vb, o[0][s]);
            o[1][s] = wmma_bf16(pa[1], vb, o[1][s]);
        }
        __syncthreads();
    }

    // ------------- finalize: X[b][q][h*64+d] = O / l  (bf16) -------------
#pragma unroll
    for (int g = 0; g < 2; ++g) {
#pragma unroll
        for (int r = 0; r < 8; ++r) {
            const float inv = 1.0f / lst[g][r];
            const int   q   = qrow + g * 16 + r + 8 * half;
            unsigned short* xr = Xo + ((size_t)b * SEQ + q) * CDIM + h * DH;
#pragma unroll
            for (int s = 0; s < 4; ++s) xr[s * 16 + lid] = f2bf(o[g][s][r] * inv);
        }
    }
}

// ---------------------------------------------------------------------------
// Host-side launcher
// ---------------------------------------------------------------------------
extern "C" void kernel_launch(void* const* d_in, const int* in_sizes, int n_in,
                              void* d_out, int out_size, void* d_ws, size_t ws_size,
                              hipStream_t stream)
{
    (void)in_sizes; (void)n_in; (void)out_size; (void)ws_size;
    const float* query = (const float*)d_in[0];
    const float* key   = (const float*)d_in[1];
    const float* value = (const float*)d_in[2];
    const float* Wq    = (const float*)d_in[3];
    const float* Wk    = (const float*)d_in[4];
    const float* Wv    = (const float*)d_in[5];
    const float* Wo    = (const float*)d_in[6];
    const float* bo    = (const float*)d_in[7];

    // workspace layout (bf16 ushorts); Xo aliases Xq (dead after Q projection)
    const size_t HE = (size_t)4 * SEQ * CDIM;             // 6,291,456 elements
    const size_t WE = (size_t)CDIM * CDIM;                //   589,824 elements
    unsigned short* Xq  = (unsigned short*)d_ws;          // also Xo later
    unsigned short* Xk  = Xq  + HE;
    unsigned short* Xv  = Xk  + HE;
    unsigned short* Qb  = Xv  + HE;
    unsigned short* Kb  = Qb  + HE;
    unsigned short* Vt  = Kb  + HE;
    unsigned short* Wqb = Vt  + HE;
    unsigned short* Wkb = Wqb + WE;
    unsigned short* Wvb = Wkb + WE;
    unsigned short* Wob = Wvb + WE;
    unsigned short* Xo  = Xq;                             // alias

    const dim3 blk(128);
    const dim3 gProj((4 * SEQ) / 128, CDIM / 64);         // (64, 12)
    const dim3 gAttn(SEQ / 128, 4 * NHEAD);               // (16, 48)
    const int  nAct4 = (int)(HE / 4);                     // 1,572,864
    const int  nW4   = (int)(WE / 4);                     //   147,456

    // one-shot bf16 conversion of activations and weights
    ca_cvt_bf16<<<nAct4 / 256, 256, 0, stream>>>(query, Xq, nAct4);
    ca_cvt_bf16<<<nAct4 / 256, 256, 0, stream>>>(key,   Xk, nAct4);
    ca_cvt_bf16<<<nAct4 / 256, 256, 0, stream>>>(value, Xv, nAct4);
    ca_cvt_bf16<<<nW4   / 256, 256, 0, stream>>>(Wq, Wqb, nW4);
    ca_cvt_bf16<<<nW4   / 256, 256, 0, stream>>>(Wk, Wkb, nW4);
    ca_cvt_bf16<<<nW4   / 256, 256, 0, stream>>>(Wv, Wvb, nW4);
    ca_cvt_bf16<<<nW4   / 256, 256, 0, stream>>>(Wo, Wob, nW4);

    // Q/K/V projections (scale Dh^-0.5 folded into Q)
    ca_proj_wmma<<<gProj, blk, 0, stream>>>(Xq, Wqb, (void*)Qb, nullptr, 0, 0.125f);
    ca_proj_wmma<<<gProj, blk, 0, stream>>>(Xk, Wkb, (void*)Kb, nullptr, 0, 1.0f);
    ca_proj_wmma<<<gProj, blk, 0, stream>>>(Xv, Wvb, (void*)Vt, nullptr, 1, 1.0f);
    // flash attention
    ca_flash_attn_wmma<<<gAttn, blk, 0, stream>>>(Qb, Kb, Vt, Xo);
    // output projection + bias
    ca_proj_wmma<<<gProj, blk, 0, stream>>>(Xo, Wob, d_out, bo, 2, 1.0f);
}
